// AccentEncoder_35562329211521
// MI455X (gfx1250) — matrix operand order repaired and checked by reading, not compiled, not run on hardware
//
#include <hip/hip_runtime.h>

// ---------------------------------------------------------------------------
// AccentEncoder on MI455X (gfx1250, wave32, WMMA + TDM).
//   conv1/conv2/convd -> implicit GEMM on v_wmma_f32_16x16x32_bf16.
//     Wave = 32x32 macro-tile (2x2 blocking, 4 f32 accumulators)
//     -> 4 WMMA per 8 b128 fragment loads. LDS im2col panel per K-section.
//   pointwise + 8-stage RVQ -> per-thread VALU with the per-stage codebook
//     staged into LDS by the Tensor Data Mover (tensor_load_to_lds,
//     TENSORcnt-tracked), instead of 5456 threads streaming 16 KB each.
// ---------------------------------------------------------------------------

typedef __attribute__((ext_vector_type(16))) __bf16        v16bf;
typedef __attribute__((ext_vector_type(8)))  float         v8f;
typedef __attribute__((ext_vector_type(4)))  unsigned int  v4u;
typedef __attribute__((ext_vector_type(4)))  unsigned int  u32x4;
typedef __attribute__((ext_vector_type(8)))  int           i32x8;
typedef __attribute__((ext_vector_type(4)))  int           i32x4;

union BfFrag {
  v4u   q[2];   // 2 x 16 bytes = 16 bf16 halves
  v16bf v;
};

__device__ __forceinline__ unsigned short f32_to_bf16(float f) {
  unsigned int u = __float_as_uint(f);
  unsigned int r = u + 0x7FFFu + ((u >> 16) & 1u);   // round-to-nearest-even
  return (unsigned short)(r >> 16);
}

static constexpr int BATCH = 16;

// ---------------------------------------------------------------------------
// Implicit-GEMM conv1d.
//   y[b,co,t] = sum_{ci,kw} w[co,ci,kw] * x[b,ci, t*STRIDE + kw - PAD]
// Block = 8 waves arranged WCO x WT; wave computes 32co x 32t (4 tiles).
// ---------------------------------------------------------------------------
template <int CIN, int KW, int STRIDE_, int PAD, int TIN, int TOUT, int COUT,
          int WCO, int WT, int KSEC, bool BN_RELU, bool IN_BF16, bool OUT_F32>
__global__ __launch_bounds__(256) void conv_wmma(
    const float* __restrict__ xin_f32,
    const unsigned short* __restrict__ xin_bf,
    const unsigned short* __restrict__ wpad,    // (COUT, KPAD) bf16, zero-padded
    const float* __restrict__ bias,
    const float* __restrict__ gamma, const float* __restrict__ beta,
    const float* __restrict__ mean,  const float* __restrict__ var,
    unsigned short* __restrict__ out_bf,
    float* __restrict__ out_f32) {

  constexpr int KTOT = CIN * KW;
  constexpr int KPAD = ((KTOT + 31) / 32) * 32;
  constexpr int NSEC = (KPAD + KSEC - 1) / KSEC;
  constexpr int PANW = WT * 32;               // t-panel width per block
  constexpr int TBLK = (TOUT + PANW - 1) / PANW;
  static_assert(WCO * 32 == COUT, "waves must cover all output channels");
  static_assert(WCO * WT == 8, "8 waves per block");
  static_assert((KSEC & (KSEC - 1)) == 0, "KSEC power of two");

  __shared__ unsigned short Bs[PANW * KSEC];  // [n][k] panel, k contiguous

  const int lane = threadIdx.x & 31;
  const int wave = threadIdx.x >> 5;
  const int nl   = lane & 15;
  const int hi   = lane >> 4;

  const int wave_co = wave % WCO;
  const int wave_t  = wave / WCO;

  const int blk   = blockIdx.x;               // grid = BATCH * TBLK, exact
  const int t_blk = blk % TBLK;
  const int b     = blk / TBLK;

  const int co0   = wave_co * 32;             // wave's 32-channel strip
  const int tpan0 = t_blk * PANW;             // panel t origin
  const int tw0   = wave_t * 32;              // wave's t offset inside panel

  // A fragments: rows (co0+nl) and (co0+16+nl); 2 contiguous 16B pieces/chunk
  const v4u* wrow0 = reinterpret_cast<const v4u*>(wpad) + (long)(co0 + nl)      * (KPAD / 8);
  const v4u* wrow1 = reinterpret_cast<const v4u*>(wpad) + (long)(co0 + 16 + nl) * (KPAD / 8);
  // B fragments: 32 contiguous bytes at [tw0(+16)+nl][k0 + hi*16]
  const v4u* bs0 = reinterpret_cast<const v4u*>(Bs) + (tw0 + nl)      * (KSEC / 8) + hi * 2;
  const v4u* bs1 = reinterpret_cast<const v4u*>(Bs) + (tw0 + 16 + nl) * (KSEC / 8) + hi * 2;

  v8f acc[2][2];
#pragma unroll
  for (int i = 0; i < 2; ++i)
#pragma unroll
    for (int j = 0; j < 2; ++j)
      acc[i][j] = v8f{0.f, 0.f, 0.f, 0.f, 0.f, 0.f, 0.f, 0.f};

  for (int sec = 0; sec < NSEC; ++sec) {
    const int ks0 = sec * KSEC;

    __syncthreads();
    // ---- cooperative im2col staging: Bs[n][kl] = patch(k = ks0+kl, col n) ----
    for (int j = threadIdx.x; j < PANW * KSEC; j += 256) {
      int n  = j / KSEC;
      int kl = j & (KSEC - 1);
      int kk = ks0 + kl;
      unsigned short v = 0;
      if (kk < KTOT) {
        int ci  = kk / KW;
        int kw  = kk - ci * KW;
        int t   = tpan0 + n;
        int tin = t * STRIDE_ + kw - PAD;
        if (t < TOUT && tin >= 0 && tin < TIN) {
          long off = ((long)b * CIN + ci) * TIN + tin;
          v = IN_BF16 ? xin_bf[off] : f32_to_bf16(xin_f32[off]);
        }
      }
      Bs[j] = v;
    }
    __syncthreads();

    // prefetch next section's weight rows while this section computes
    if (sec + 1 < NSEC) {
      __builtin_prefetch((const char*)(wrow0 + (ks0 + KSEC) / 8), 0, 3);
      __builtin_prefetch((const char*)(wrow1 + (ks0 + KSEC) / 8), 0, 3);
    }

    const int nch = ((KPAD - ks0) < KSEC ? (KPAD - ks0) : KSEC) / 32;
    const int c4  = ks0 / 8;                 // v4u index of section start
    for (int cl = 0; cl < nch; ++cl) {
      BfFrag A0, A1, B0, B1;
      A0.q[0] = wrow0[c4 + cl * 4 + hi * 2];
      A0.q[1] = wrow0[c4 + cl * 4 + hi * 2 + 1];
      A1.q[0] = wrow1[c4 + cl * 4 + hi * 2];
      A1.q[1] = wrow1[c4 + cl * 4 + hi * 2 + 1];
      B0.q[0] = bs0[cl * 4];
      B0.q[1] = bs0[cl * 4 + 1];
      B1.q[0] = bs1[cl * 4];
      B1.q[1] = bs1[cl * 4 + 1];
      acc[0][0] = __builtin_amdgcn_wmma_f32_16x16x32_bf16(
          false, A0.v, false, B0.v, (short)0, acc[0][0], false, false);
      acc[0][1] = __builtin_amdgcn_wmma_f32_16x16x32_bf16(
          false, A0.v, false, B1.v, (short)0, acc[0][1], false, false);
      acc[1][0] = __builtin_amdgcn_wmma_f32_16x16x32_bf16(
          false, A1.v, false, B0.v, (short)0, acc[1][0], false, false);
      acc[1][1] = __builtin_amdgcn_wmma_f32_16x16x32_bf16(
          false, A1.v, false, B1.v, (short)0, acc[1][1], false, false);
    }
  }

  // ---- epilogue: bias (+ BN + ReLU), store 4 tiles ----
#pragma unroll
  for (int i = 0; i < 2; ++i) {
#pragma unroll
    for (int j = 0; j < 2; ++j) {
      const int t = tpan0 + tw0 + j * 16 + nl;
#pragma unroll
      for (int r = 0; r < 8; ++r) {
        int co  = co0 + i * 16 + r + hi * 8;    // C/D layout: M = r (+8 hi half)
        float y = acc[i][j][r] + bias[co];
        if (BN_RELU) {
          float inv = rsqrtf(var[co] + 1e-5f);
          y = (y - mean[co]) * (inv * gamma[co]) + beta[co];
          y = fmaxf(y, 0.f);
        }
        if (t < TOUT) {
          long off = ((long)b * COUT + co) * TOUT + t;
          if (OUT_F32) out_f32[off] = y;
          else         out_bf[off]  = f32_to_bf16(y);
        }
      }
    }
  }
}

// ---------------------------------------------------------------------------
// Pack f32 weights (COUT x KTOT) -> zero-padded bf16 (COUT x KPAD).
// ---------------------------------------------------------------------------
__global__ void pack_wpad(const float* __restrict__ src,
                          unsigned short* __restrict__ dst,
                          int ktot, int kpad, int n /* = COUT*KPAD */) {
  int i = blockIdx.x * blockDim.x + threadIdx.x;
  if (i >= n) return;
  int co = i / kpad;
  int k  = i - co * kpad;
  dst[i] = (k < ktot) ? f32_to_bf16(src[(long)co * ktot + k]) : (unsigned short)0;
}

// ---------------------------------------------------------------------------
// Pointwise 128->16 conv + 8-stage residual VQ. One thread per (b, t).
// Per stage, the 256x16 f32 codebook (16 KB) is DMA'd into LDS by the TDM
// (tensor_load_to_lds, TENSORcnt) and the argmin search reads LDS.
// ---------------------------------------------------------------------------
__global__ __launch_bounds__(256) void pw_rvq(
    const float* __restrict__ e,       // (B,128,TP) = e_a_unq (in d_out)
    const float* __restrict__ wp,      // (16,128)
    const float* __restrict__ bp,      // (16,)
    const float* __restrict__ cb,      // (8,256,16)
    float* __restrict__ out_idx,       // (8,B,TP) as float
    int TP) {
  __shared__ float cbs[256 * 16];      // sole LDS object -> offset 0

  const int i      = blockIdx.x * blockDim.x + threadIdx.x;
  const bool active = i < BATCH * TP;
  const int ii = active ? i : 0;       // clamp so inactive threads stay in-bounds
  const int b = ii / TP, t = ii - b * TP;

  float r[16];
#pragma unroll
  for (int j = 0; j < 16; ++j) r[j] = bp[j];
  for (int c = 0; c < 128; ++c) {
    float xv = e[((long)b * 128 + c) * TP + t];
#pragma unroll
    for (int j = 0; j < 16; ++j) r[j] = fmaf(wp[j * 128 + c], xv, r[j]);
  }

  for (int q = 0; q < 8; ++q) {
    __syncthreads();                       // everyone done with previous cbs
    if ((threadIdx.x >> 5) == 0) {         // wave 0 issues the TDM descriptor
      unsigned long long ga =
          (unsigned long long)(const void*)(cb + (long)q * 256 * 16);
      // D# group 0: count=1 | lds_addr=0 | global_addr[56:0] | type=2
      u32x4 g0 = {1u, 0u, (unsigned)ga,
                  (unsigned)((ga >> 32) & 0x01FFFFFFull) | (2u << 30)};
      // D# group 1: data_size=4B; tensor 1024x4, tile 1024x4, stride0=1024
      //   (contiguous 16 KB 2-D tile)
      i32x8 g1 = {(int)(2u << 16),          // data_size=2 (4 bytes)
                  (int)(1024u << 16),       // tensor_dim0[15:0]=1024 @ [63:48]
                  (int)(4u << 16),          // dim0[31:16]=0, tensor_dim1=4
                  (int)(1024u << 16),       // dim1[31:16]=0, tile_dim0=1024
                  4,                        // tile_dim1=4, tile_dim2=0
                  1024,                     // tensor_dim0_stride[31:0]
                  0, 0};
      i32x4 z4 = {0, 0, 0, 0};
      i32x8 z8 = {0, 0, 0, 0, 0, 0, 0, 0};
      __builtin_amdgcn_tensor_load_to_lds(g0, g1, z4, z4, z8, 0);
      __builtin_amdgcn_s_wait_tensorcnt(0);
    }
    __syncthreads();                       // staged codebook visible

    int best = 0;
    float bestd = 3.402823e38f;
    for (int k = 0; k < 256; ++k) {
      const float* cw = cbs + k * 16;      // same addr all lanes -> broadcast
      float d = 0.f;
#pragma unroll
      for (int j = 0; j < 16; ++j) {
        float df = r[j] - cw[j];
        d = fmaf(df, df, d);
      }
      if (d < bestd) { bestd = d; best = k; }   // first-min tie-break == argmin
    }
    if (active)
      out_idx[((long)q * BATCH + b) * TP + t] = (float)best;
    const float* cw = cbs + best * 16;
#pragma unroll
    for (int j = 0; j < 16; ++j) r[j] -= cw[j];
  }
}

// ---------------------------------------------------------------------------
// Launch
// ---------------------------------------------------------------------------
extern "C" void kernel_launch(void* const* d_in, const int* in_sizes, int n_in,
                              void* d_out, int out_size, void* d_ws, size_t ws_size,
                              hipStream_t stream) {
  (void)in_sizes; (void)n_in; (void)out_size; (void)ws_size;

  const float* mel = (const float*)d_in[0];
  const float* w1  = (const float*)d_in[1];
  const float* b1  = (const float*)d_in[2];
  const float* g1  = (const float*)d_in[3];
  const float* be1 = (const float*)d_in[4];
  const float* m1  = (const float*)d_in[5];
  const float* v1  = (const float*)d_in[6];
  const float* w2  = (const float*)d_in[7];
  const float* b2  = (const float*)d_in[8];
  const float* g2  = (const float*)d_in[9];
  const float* be2 = (const float*)d_in[10];
  const float* m2  = (const float*)d_in[11];
  const float* v2  = (const float*)d_in[12];
  const float* wd  = (const float*)d_in[13];
  const float* bd  = (const float*)d_in[14];
  const float* wp  = (const float*)d_in[15];
  const float* bp  = (const float*)d_in[16];
  const float* cb  = (const float*)d_in[17];

  constexpr int T  = 4096;
  constexpr int TP = 341;                       // floor((4096+12-24)/12)+1

  constexpr int K1 = 80 * 5,    KP1 = ((K1 + 31) / 32) * 32;   // 400 -> 416
  constexpr int K2 = 256 * 5,   KP2 = ((K2 + 31) / 32) * 32;   // 1280
  constexpr int KD = 256 * 24,  KPD = ((KD + 31) / 32) * 32;   // 6144

  constexpr long NW1 = 256L * KP1;              // padded bf16 weight elems
  constexpr long NW2 = 256L * KP2;
  constexpr long NWD = 128L * KPD;
  constexpr long NX  = (long)BATCH * 256 * T;   // intermediate activations

  // workspace layout (bytes, 256-aligned)
  constexpr size_t OFF_W1 = 0;
  constexpr size_t OFF_W2 = OFF_W1 + ((NW1 * 2 + 255) & ~255UL);
  constexpr size_t OFF_WD = OFF_W2 + ((NW2 * 2 + 255) & ~255UL);
  constexpr size_t OFF_X1 = OFF_WD + ((NWD * 2 + 255) & ~255UL);
  constexpr size_t OFF_X2 = OFF_X1 + ((NX * 2 + 255) & ~255UL);

  char* ws = (char*)d_ws;
  unsigned short* w1b = (unsigned short*)(ws + OFF_W1);
  unsigned short* w2b = (unsigned short*)(ws + OFF_W2);
  unsigned short* wdb = (unsigned short*)(ws + OFF_WD);
  unsigned short* x1b = (unsigned short*)(ws + OFF_X1);
  unsigned short* x2b = (unsigned short*)(ws + OFF_X2);

  float* e_a_unq = (float*)d_out;                          // (16,128,341)
  float* idx_out = (float*)d_out + (long)BATCH * 128 * TP; // (8,16,341)

  // 1) pack + zero-pad weights to bf16
  pack_wpad<<<(int)((NW1 + 255) / 256), 256, 0, stream>>>(w1, w1b, K1, KP1, (int)NW1);
  pack_wpad<<<(int)((NW2 + 255) / 256), 256, 0, stream>>>(w2, w2b, K2, KP2, (int)NW2);
  pack_wpad<<<(int)((NWD + 255) / 256), 256, 0, stream>>>(wd, wdb, KD, KPD, (int)NWD);

  // 2) conv1 (80->256, k5, p2) + BN + ReLU -> x1 (bf16)
  //    block = 8x1 waves = 256co x 32t; blocks = 16 * (4096/32) = 2048
  conv_wmma<80, 5, 1, 2, T, T, 256, 8, 1, 256, true, false, false>
      <<<2048, 256, 0, stream>>>(mel, nullptr, w1b, b1, g1, be1, m1, v1,
                                 x1b, nullptr);

  // 3) conv2 (256->256, k5, p2) + BN + ReLU -> x2 (bf16)
  conv_wmma<256, 5, 1, 2, T, T, 256, 8, 1, 256, true, true, false>
      <<<2048, 256, 0, stream>>>(nullptr, x1b, w2b, b2, g2, be2, m2, v2,
                                 x2b, nullptr);

  // 4) downsample conv (256->128, k24, s12, p6) + bias -> e_a_unq (f32, d_out)
  //    block = 4x2 waves = 128co x 64t; blocks = 16 * ceil(341/64) = 96
  conv_wmma<256, 24, 12, 6, T, TP, 128, 4, 2, 256, false, true, true>
      <<<96, 256, 0, stream>>>(nullptr, x2b, wdb, bd,
                               nullptr, nullptr, nullptr, nullptr,
                               nullptr, e_a_unq);

  // 5) pointwise 128->16 + 8-stage RVQ (TDM-staged codebooks) -> indices
  pw_rvq<<<(BATCH * TP + 255) / 256, 256, 0, stream>>>(
      e_a_unq, wp, bp, cb, idx_out, TP);
}
